// DirectRegressionHead_46729244180888
// MI455X (gfx1250) — matrix operand order repaired and checked by reading, not compiled, not run on hardware
//
#include <hip/hip_runtime.h>

#define HH    256
#define GG    768
#define CTXD  128
#define PRED  12
#define WAVES 4
#define MROWS 16

typedef __attribute__((ext_vector_type(16))) __bf16         v16bf;
typedef __attribute__((ext_vector_type(8)))  float          v8f;
typedef __attribute__((ext_vector_type(8)))  unsigned short us8;

union FragU { v16bf v; us8 h[2]; unsigned short s[16]; };

// ---------- scalar helpers ----------
static __device__ __forceinline__ unsigned short f2bf(float f){
  unsigned x = __float_as_uint(f);
  unsigned r = ((x >> 16) & 1u) + 0x7fffu;   // round-to-nearest-even
  return (unsigned short)((x + r) >> 16);
}
static __device__ __forceinline__ float bf2f(unsigned short u){
  return __uint_as_float(((unsigned)u) << 16);
}
static __device__ __forceinline__ v8f splat8(float x){
  v8f r;
#pragma unroll
  for (int i = 0; i < 8; ++i) r[i] = x;
  return r;
}
static __device__ __forceinline__ float sigm(float x){
  return 1.f / (1.f + __expf(-x));
}
static __device__ __forceinline__ float tanh_f(float x){
  float e = __expf(-2.f * x);
  return (1.f - e) / (1.f + e);
}
static __device__ __forceinline__ float gelu_f(float x){
  return 0.5f * x * (1.f + erff(x * 0.70710678118654752f));
}
static __device__ __forceinline__ v8f wmma_bf16(v16bf a, v16bf b, v8f c){
  // D = A(16x32 bf16) x B(32x16 bf16) + C(16x16 f32)
  return __builtin_amdgcn_wmma_f32_16x16x32_bf16(false, a, false, b, (short)0, c, false, false);
}

// ---------- fragment loaders ----------
// A fragment (16x32 bf16) from LDS, row-major [16][stride].
// Layout (ISA 7.12.2): lane<16: row=lane, K runs {kt*32+0..7, kt*32+16..23};
// lane>=16: row=lane-16, runs shifted by +8.
static __device__ __forceinline__ v16bf lds_afrag(const unsigned short* p, int stride, int kt, int lane){
  int row = lane & 15;
  int kh  = (lane & 16) ? 8 : 0;
  const unsigned short* q = p + row * stride + kt * 32 + kh;
  FragU u;
  u.h[0] = *(const us8*)(q);        // 16B -> ds_load_b128
  u.h[1] = *(const us8*)(q + 16);
  return u.v;
}
// B fragment (32x16 bf16) from global bf16 weight, row-major [N][Kpad] (gi = x @ W^T).
// lane<16: col n0+lane, K = {kt*32+0..7, kt*32+16..23} ; lane>=16: same col, runs shifted by +8.
static __device__ __forceinline__ v16bf gl_bfrag(const unsigned short* __restrict__ W,
                                                 int Kpad, int n0, int kt, int lane){
  int n  = lane & 15;
  int ks = (lane & 16) ? 8 : 0;
  const unsigned short* q = W + (size_t)(n0 + n) * Kpad + kt * 32 + ks;
  FragU u;
  u.h[0] = *(const us8*)(q);        // 16B -> global_load_b128
  u.h[1] = *(const us8*)(q + 16);
  return u.v;
}
// A fragment built from global f32 rows (ctx matrix)
static __device__ v16bf ctx_afrag(const float* __restrict__ ctx, int rowbase, int kt, int lane){
  FragU u;
  int row = lane & 15;
  int kh  = (lane & 16) ? 8 : 0;
  const float* p = ctx + (size_t)(rowbase + row) * CTXD + kt * 32 + kh;
#pragma unroll
  for (int e = 0; e < 8; ++e) u.s[e]     = f2bf(p[e]);
#pragma unroll
  for (int e = 0; e < 8; ++e) u.s[8 + e] = f2bf(p[16 + e]);
  return u.v;
}

// ---------- one GRU cell step for a 16-row tile ----------
// gates: r = sigmoid(gi_r+gh_r), z = sigmoid(gi_z+gh_z), n = tanh(gi_n + r*gh_n)
// h' = (1-z)*n + z*h     (PyTorch gate order r,z,n; rows 0/256/512 of 768-row weights)
// XKT = number of x K-tiles to CACHE in registers (0 => stream or none);
// XSTREAM_KT = number of x K-tiles to stream from LDS each nt (layer-1 path).
template<int XKT, int XSTREAM_KT, bool XF>
static __device__ void gru_cell_t(const unsigned short* __restrict__ hcur,
                                  unsigned short* __restrict__ hnext,
                                  const unsigned short* __restrict__ xlds, int xstride,
                                  const v16bf* xfrag, int xfkt,
                                  const unsigned short* __restrict__ Wih, int KpadX,
                                  const unsigned short* __restrict__ Whh,
                                  const float* __restrict__ bih,
                                  const float* __restrict__ bhh, int lane)
{
  const int cl = lane & 15;
  const int rs = (lane >> 4) & 1;

  // cache invariant A-fragments in registers (reused across all 16 nt tiles)
  v16bf ah_[8];
#pragma unroll
  for (int kt = 0; kt < 8; ++kt) ah_[kt] = lds_afrag(hcur, HH, kt, lane);
  v16bf ax_[XKT > 0 ? XKT : 1];
#pragma unroll
  for (int kt = 0; kt < XKT; ++kt) ax_[kt] = lds_afrag(xlds, xstride, kt, lane);

  for (int nt = 0; nt < 16; ++nt){
    const int c  = nt * 16 + cl;
    const int n0 = nt * 16;
    v8f ar = splat8(bih[c]          + bhh[c]);
    v8f az = splat8(bih[HH + c]     + bhh[HH + c]);
    v8f ai = splat8(bih[2 * HH + c]);               // x-part of n gate (+bias)
    v8f ah = splat8(bhh[2 * HH + c]);               // h-part of n gate (+bias)
#pragma unroll
    for (int kt = 0; kt < XKT; ++kt){
      ar = wmma_bf16(ax_[kt], gl_bfrag(Wih, KpadX, n0,          kt, lane), ar);
      az = wmma_bf16(ax_[kt], gl_bfrag(Wih, KpadX, HH + n0,     kt, lane), az);
      ai = wmma_bf16(ax_[kt], gl_bfrag(Wih, KpadX, 2 * HH + n0, kt, lane), ai);
    }
#pragma unroll
    for (int kt = 0; kt < XSTREAM_KT; ++kt){
      v16bf a = lds_afrag(xlds, xstride, kt, lane);
      ar = wmma_bf16(a, gl_bfrag(Wih, KpadX, n0,          kt, lane), ar);
      az = wmma_bf16(a, gl_bfrag(Wih, KpadX, HH + n0,     kt, lane), az);
      ai = wmma_bf16(a, gl_bfrag(Wih, KpadX, 2 * HH + n0, kt, lane), ai);
    }
    if (XF){
      v16bf a = *xfrag;
      ar = wmma_bf16(a, gl_bfrag(Wih, KpadX, n0,          xfkt, lane), ar);
      az = wmma_bf16(a, gl_bfrag(Wih, KpadX, HH + n0,     xfkt, lane), az);
      ai = wmma_bf16(a, gl_bfrag(Wih, KpadX, 2 * HH + n0, xfkt, lane), ai);
    }
#pragma unroll
    for (int kt = 0; kt < 8; ++kt){
      ar = wmma_bf16(ah_[kt], gl_bfrag(Whh, HH, n0,          kt, lane), ar);
      az = wmma_bf16(ah_[kt], gl_bfrag(Whh, HH, HH + n0,     kt, lane), az);
      ah = wmma_bf16(ah_[kt], gl_bfrag(Whh, HH, 2 * HH + n0, kt, lane), ah);
    }
#pragma unroll
    for (int j = 0; j < 8; ++j){
      float r  = sigm(ar[j]);
      float z  = sigm(az[j]);
      float nn = tanh_f(ai[j] + r * ah[j]);
      int row  = j + rs * 8;                        // C-layout row mapping
      float hold = bf2f(hcur[row * HH + c]);
      hnext[row * HH + c] = f2bf((1.f - z) * nn + z * hold);
    }
  }
}

// ---------- encoder input fragment (K valid 0..5: vel, me_vel, acc) ----------
static __device__ v16bf enc_frag(const float* __restrict__ traj, const float* __restrict__ me,
                                 int Bn, int rowbase, int t, int lane){
  FragU u;
#pragma unroll
  for (int i = 0; i < 16; ++i) u.s[i] = 0;
  if (lane < 16){
    int r = rowbase + lane;
    auto TT = [&](int tt, int d){ return traj[((size_t)tt * Bn + r) * 2 + d]; };
    auto MM = [&](int tt, int d){ return me  [((size_t)tt * Bn + r) * 2 + d]; };
    float vx = TT(t + 1, 0) - TT(t, 0);
    float vy = TT(t + 1, 1) - TT(t, 1);
    float mx = MM(t + 1, 0) - MM(t, 0);
    float my = MM(t + 1, 1) - MM(t, 1);
    int ta = (t < 1) ? 1 : t;                       // acc[0] duplicated from acc[1]
    float ax = TT(ta + 1, 0) - 2.f * TT(ta, 0) + TT(ta - 1, 0);
    float ay = TT(ta + 1, 1) - 2.f * TT(ta, 1) + TT(ta - 1, 1);
    u.s[0] = f2bf(vx); u.s[1] = f2bf(vy);
    u.s[2] = f2bf(mx); u.s[3] = f2bf(my);
    u.s[4] = f2bf(ax); u.s[5] = f2bf(ay);
  }
  return u.v;
}
// decoder initial carry: (last_vel, last_acc, last_me) at K=128..133
static __device__ v16bf dyn_init(const float* __restrict__ traj, const float* __restrict__ me,
                                 int Bn, int rowbase, int lane){
  FragU u;
#pragma unroll
  for (int i = 0; i < 16; ++i) u.s[i] = 0;
  if (lane < 16){
    int r = rowbase + lane;
    auto TT = [&](int tt, int d){ return traj[((size_t)tt * Bn + r) * 2 + d]; };
    auto MM = [&](int tt, int d){ return me  [((size_t)tt * Bn + r) * 2 + d]; };
    u.s[0] = f2bf(TT(7,0) - TT(6,0));
    u.s[1] = f2bf(TT(7,1) - TT(6,1));
    u.s[2] = f2bf(TT(7,0) - 2.f*TT(6,0) + TT(5,0));
    u.s[3] = f2bf(TT(7,1) - 2.f*TT(6,1) + TT(5,1));
    u.s[4] = f2bf(MM(7,0) - MM(6,0));
    u.s[5] = f2bf(MM(7,1) - MM(6,1));
  }
  return u.v;
}

// ---------- weight convert/pad f32 -> bf16 ----------
__global__ void cvt_pad(const float* __restrict__ src, unsigned short* __restrict__ dst,
                        int Ns, int Ks, int Nd, int Kd){
  int i = blockIdx.x * blockDim.x + threadIdx.x;
  int total = Nd * Kd;
  if (i >= total) return;
  int n = i / Kd, k = i % Kd;
  float v = (n < Ns && k < Ks) ? src[(size_t)n * Ks + k] : 0.f;
  unsigned x = __float_as_uint(v);
  unsigned r = ((x >> 16) & 1u) + 0x7fffu;
  dst[i] = (unsigned short)((x + r) >> 16);
}

// ---------- fused persistent kernel: encoder + decoder + projections ----------
__global__ __launch_bounds__(WAVES * 32) void trajgru_fused(
    const float* __restrict__ ctx, const float* __restrict__ traj, const float* __restrict__ me,
    const float* __restrict__ bih0, const float* __restrict__ bhh0,
    const float* __restrict__ bih1, const float* __restrict__ bhh1,
    const float* __restrict__ bihs, const float* __restrict__ bhhs,
    const float* __restrict__ p1b,  const float* __restrict__ p2b,
    const float* __restrict__ a1b,  const float* __restrict__ a2b,
    const float* __restrict__ cb,
    const unsigned short* __restrict__ Wih0, const unsigned short* __restrict__ Whh0,
    const unsigned short* __restrict__ Wih1, const unsigned short* __restrict__ Whh1,
    const unsigned short* __restrict__ Wihs, const unsigned short* __restrict__ Whhs,
    const unsigned short* __restrict__ Cwp,  const unsigned short* __restrict__ P1w,
    const unsigned short* __restrict__ P2w,  const unsigned short* __restrict__ A1w,
    const unsigned short* __restrict__ A2w,
    float* __restrict__ out, int Bn)
{
  __shared__ __align__(16) unsigned short smem[WAVES * 16384]; // 32 KB per wave
  const int lane = threadIdx.x & 31;
  const int w    = threadIdx.x >> 5;
  unsigned short* base = smem + w * 16384;
  unsigned short* R0 = base;            // 16x256 bf16 (8KB) regions
  unsigned short* R1 = base + 4096;
  unsigned short* R2 = base + 8192;
  unsigned short* R3 = base + 12288;
  const int rowbase = blockIdx.x * (WAVES * MROWS) + w * MROWS;
  const int cl = lane & 15;
  const int rs = (lane >> 4) & 1;

  // zero initial hidden states h0, h1
  for (int i = lane; i < 2048; i += 32){
    ((unsigned*)R0)[i] = 0u;
    ((unsigned*)R2)[i] = 0u;
  }

  unsigned short *h0c = R0, *h0n = R1, *h1c = R2, *h1n = R3;

  // ---- velocity encoder: 2-layer GRU over 7 steps (per-wave, no barriers) ----
  for (int t = 0; t < 7; ++t){
    v16bf xf = enc_frag(traj, me, Bn, rowbase, t, lane);
    gru_cell_t<0, 0, true >(h0c, h0n, nullptr, 0, &xf, 0, Wih0, 32, Whh0, bih0, bhh0, lane);
    gru_cell_t<0, 8, false>(h1c, h1n, h0n, HH, nullptr, 0, Wih1, HH, Whh1, bih1, bhh1, lane);
    unsigned short* tp;
    tp = h0c; h0c = h0n; h0n = tp;
    tp = h1c; h1c = h1n; h1n = tp;
  }

  // ---- decoder scratch (re-use freed h0 regions) ----
  unsigned short* ctxp = R0;            // 16 x 128 bf16
  unsigned short* g1   = R0 + 2048;     // 16 x 128 bf16
  unsigned short* pa   = R1;            // 16 x  64 bf16
  unsigned short* dyn  = R1 + 1024;     // 16 x   8 bf16 (cv,ca,cm shuffle buffer)

  // ctx_p = ctx @ Cw^T + cb (computed once, kept bf16 in LDS)
  {
    v16bf ac_[4];
#pragma unroll
    for (int kt = 0; kt < 4; ++kt) ac_[kt] = ctx_afrag(ctx, rowbase, kt, lane);
    for (int nt = 0; nt < 8; ++nt){
      v8f acc = splat8(cb[nt * 16 + cl]);
#pragma unroll
      for (int kt = 0; kt < 4; ++kt)
        acc = wmma_bf16(ac_[kt], gl_bfrag(Cwp, CTXD, nt * 16, kt, lane), acc);
#pragma unroll
      for (int j = 0; j < 8; ++j)
        ctxp[(j + rs * 8) * CTXD + nt * 16 + cl] = f2bf(acc[j]);
    }
  }

  v16bf dynf = dyn_init(traj, me, Bn, rowbase, lane);
  unsigned short *hc = h1c, *hn = h1n;

  for (int s = 0; s < PRED; ++s){
    // step_rnn: input = [ctx_p(128) | cv,ca,cm(6) | pad..160]
    gru_cell_t<4, 0, true>(hc, hn, ctxp, CTXD, &dynf, 4, Wihs, 160, Whhs, bihs, bhhs, lane);
    unsigned short* tp = hc; hc = hn; hn = tp;

    // cache new-h A-fragments once for both projection GEMMs
    v16bf ph[8];
#pragma unroll
    for (int kt = 0; kt < 8; ++kt) ph[kt] = lds_afrag(hc, HH, kt, lane);

    // step_proj: gelu(h @ P1^T + p1b) -> g1
    for (int nt = 0; nt < 8; ++nt){
      v8f acc = splat8(p1b[nt * 16 + cl]);
#pragma unroll
      for (int kt = 0; kt < 8; ++kt)
        acc = wmma_bf16(ph[kt], gl_bfrag(P1w, HH, nt * 16, kt, lane), acc);
#pragma unroll
      for (int j = 0; j < 8; ++j)
        g1[(j + rs * 8) * CTXD + nt * 16 + cl] = f2bf(gelu_f(acc[j]));
    }
    // delta = g1 @ P2^T + p2b  (cols 0..3 valid; weights zero-padded to N=16)
    v8f dacc = splat8(cl < 4 ? p2b[cl] : 0.f);
#pragma unroll
    for (int kt = 0; kt < 4; ++kt){
      v16bf a = lds_afrag(g1, CTXD, kt, lane);
      dacc = wmma_bf16(a, gl_bfrag(P2w, CTXD, 0, kt, lane), dacc);
    }
    // accel_proj: gelu(h @ A1^T + a1b) -> pa ; acorr = pa @ A2^T + a2b
    for (int nt = 0; nt < 4; ++nt){
      v8f acc = splat8(a1b[nt * 16 + cl]);
#pragma unroll
      for (int kt = 0; kt < 8; ++kt)
        acc = wmma_bf16(ph[kt], gl_bfrag(A1w, HH, nt * 16, kt, lane), acc);
#pragma unroll
      for (int j = 0; j < 8; ++j)
        pa[(j + rs * 8) * 64 + nt * 16 + cl] = f2bf(gelu_f(acc[j]));
    }
    v8f aacc = splat8(cl < 2 ? a2b[cl] : 0.f);
#pragma unroll
    for (int kt = 0; kt < 2; ++kt){
      v16bf a = lds_afrag(pa, 64, kt, lane);
      aacc = wmma_bf16(a, gl_bfrag(A2w, 64, 0, kt, lane), aacc);
    }

    // emit delta -> out[B, 12, 4]
    if (cl < 4){
#pragma unroll
      for (int j = 0; j < 8; ++j){
        int row = rowbase + j + rs * 8;
        out[(size_t)row * (PRED * 4) + s * 4 + cl] = dacc[j];
      }
    }
    // carry update (cv=delta[:,0:2], ca=acorr, cm=delta[:,2:4]) via LDS lane shuffle
    if (cl < 2){
#pragma unroll
      for (int j = 0; j < 8; ++j){
        dyn[(j + rs * 8) * 8 + cl]     = f2bf(dacc[j]);   // cv
        dyn[(j + rs * 8) * 8 + 2 + cl] = f2bf(aacc[j]);   // ca
      }
    } else if (cl < 4){
#pragma unroll
      for (int j = 0; j < 8; ++j)
        dyn[(j + rs * 8) * 8 + 2 + cl] = f2bf(dacc[j]);   // cm (idx 4,5)
    }
    FragU u;
#pragma unroll
    for (int i = 0; i < 16; ++i) u.s[i] = 0;
    if (lane < 16){
#pragma unroll
      for (int i = 0; i < 6; ++i) u.s[i] = dyn[lane * 8 + i];
    }
    dynf = u.v;
  }
}

// ---------- workspace layout (bf16 element offsets) ----------
static constexpr size_t O_WIH0 = 0;
static constexpr size_t O_WHH0 = O_WIH0 + (size_t)GG * 32;
static constexpr size_t O_WIH1 = O_WHH0 + (size_t)GG * HH;
static constexpr size_t O_WHH1 = O_WIH1 + (size_t)GG * HH;
static constexpr size_t O_WIHS = O_WHH1 + (size_t)GG * HH;
static constexpr size_t O_WHHS = O_WIHS + (size_t)GG * 160;
static constexpr size_t O_CW   = O_WHHS + (size_t)GG * HH;
static constexpr size_t O_P1   = O_CW   + (size_t)CTXD * CTXD;
static constexpr size_t O_P2   = O_P1   + (size_t)CTXD * HH;
static constexpr size_t O_A1   = O_P2   + (size_t)16 * CTXD;
static constexpr size_t O_A2   = O_A1   + (size_t)64 * HH;

extern "C" void kernel_launch(void* const* d_in, const int* in_sizes, int n_in,
                              void* d_out, int out_size, void* d_ws, size_t ws_size,
                              hipStream_t stream)
{
  const float* ctx   = (const float*)d_in[0];
  const float* traj  = (const float*)d_in[1];
  const float* me    = (const float*)d_in[2];
  const float* Wih0f = (const float*)d_in[3];
  const float* Whh0f = (const float*)d_in[4];
  const float* bih0  = (const float*)d_in[5];
  const float* bhh0  = (const float*)d_in[6];
  const float* Wih1f = (const float*)d_in[7];
  const float* Whh1f = (const float*)d_in[8];
  const float* bih1  = (const float*)d_in[9];
  const float* bhh1  = (const float*)d_in[10];
  const float* Wihsf = (const float*)d_in[11];
  const float* Whhsf = (const float*)d_in[12];
  const float* bihs  = (const float*)d_in[13];
  const float* bhhs  = (const float*)d_in[14];
  const float* P1f   = (const float*)d_in[15];
  const float* p1b   = (const float*)d_in[16];
  const float* P2f   = (const float*)d_in[17];
  const float* p2b   = (const float*)d_in[18];
  const float* A1f   = (const float*)d_in[19];
  const float* a1b   = (const float*)d_in[20];
  const float* A2f   = (const float*)d_in[21];
  const float* a2b   = (const float*)d_in[22];
  const float* Cwf   = (const float*)d_in[23];
  const float* cb    = (const float*)d_in[24];

  unsigned short* ws = (unsigned short*)d_ws;
  auto conv = [&](const float* s, size_t off, int Ns, int Ks, int Nd, int Kd){
    int total = Nd * Kd;
    cvt_pad<<<(total + 255) / 256, 256, 0, stream>>>(s, ws + off, Ns, Ks, Nd, Kd);
  };
  conv(Wih0f, O_WIH0, GG, 6,    GG, 32);
  conv(Whh0f, O_WHH0, GG, HH,   GG, HH);
  conv(Wih1f, O_WIH1, GG, HH,   GG, HH);
  conv(Whh1f, O_WHH1, GG, HH,   GG, HH);
  conv(Wihsf, O_WIHS, GG, 134,  GG, 160);
  conv(Whhsf, O_WHHS, GG, HH,   GG, HH);
  conv(Cwf,   O_CW,   CTXD, CTXD, CTXD, CTXD);
  conv(P1f,   O_P1,   CTXD, HH,   CTXD, HH);
  conv(P2f,   O_P2,   4,  CTXD,   16,  CTXD);
  conv(A1f,   O_A1,   64, HH,     64,  HH);
  conv(A2f,   O_A2,   2,  64,     16,  64);

  int Bn   = in_sizes[0] / CTXD;            // 32768
  int grid = Bn / (WAVES * MROWS);          // 512 blocks of 4 wave32s
  trajgru_fused<<<grid, WAVES * 32, 0, stream>>>(
      ctx, traj, me,
      bih0, bhh0, bih1, bhh1, bihs, bhhs,
      p1b, p2b, a1b, a2b, cb,
      ws + O_WIH0, ws + O_WHH0, ws + O_WIH1, ws + O_WHH1,
      ws + O_WIHS, ws + O_WHHS, ws + O_CW,  ws + O_P1,
      ws + O_P2,   ws + O_A1,   ws + O_A2,
      (float*)d_out, Bn);
}
